// ADIAModel_60756607369347
// MI455X (gfx1250) — matrix operand order repaired and checked by reading, not compiled, not run on hardware
//
#include <hip/hip_runtime.h>
#include <hip/hip_bf16.h>

typedef __attribute__((ext_vector_type(16))) _Float16 v16h;
typedef __attribute__((ext_vector_type(8)))  float    v8f;

#define B_SZ    16
#define E_EDGES 132
#define C_INCH  8
#define N_LEN   512
#define D_MODEL 64
#define H_HEADS 4
#define M_TOT   (B_SZ * E_EDGES)   // 2112 sequences

__device__ __forceinline__ float gelu_exact(float x) {
  return 0.5f * x * (1.0f + erff(x * 0.70710678118654752f));
}
__device__ __forceinline__ float wred_sum(float v) {
#pragma unroll
  for (int m = 16; m > 0; m >>= 1) v += __shfl_xor(v, m, 32);
  return v;
}
__device__ __forceinline__ float wred_max(float v) {
#pragma unroll
  for (int m = 16; m > 0; m >>= 1) v = fmaxf(v, __shfl_xor(v, m, 32));
  return v;
}

struct ConvP { const float* w; const float* g; const float* b; };
struct K1Args { ConvP cp[5]; };

// ---------------------------------------------------------------------------
// K1: fused stem -> 5x (conv1d(64,64,3) as WMMA GEMM + GroupNorm + GELU +
// residual) -> avg pool.  One workgroup (8 wave32) per sequence m.
// Activations live in LDS as xT[n][c] (f16, with 1-row zero halo each side).
// GEMM: y[64x512] = W[64x192] * im2col(x)[192x512], K = t*64 + c.
// ---------------------------------------------------------------------------
__global__ __launch_bounds__(256)
void k1_stem_conv_pool(const float* __restrict__ edge_data,
                       const float* __restrict__ stem_w,
                       const float* __restrict__ stem_b,
                       K1Args args,
                       float* __restrict__ conv_emb)
{
  __shared__ __align__(16) _Float16 xT[514 * 64];   // halo rows 0 and 513
  __shared__ __align__(16) _Float16 yT[512 * 64];
  __shared__ __align__(16) _Float16 wA[64 * 192];   // repacked weights [d][k]
  __shared__ float gG[64], gB[64];
  __shared__ float gstat[16];
  __shared__ float pool[256];

  const int m    = blockIdx.x;
  const int tid  = threadIdx.x;
  const int lane = tid & 31;
  const int w    = tid >> 5;
  const float* in = edge_data + (size_t)m * (C_INCH * N_LEN);

  // ---- Stem: x[d][n] = sum_c in[c][n] * Ws[c][d] + bs[d] ----
  for (int idx = tid; idx < N_LEN * 64; idx += 256) {
    int n = idx >> 6, d = idx & 63;
    float s = stem_b[d];
#pragma unroll
    for (int c = 0; c < C_INCH; ++c) s += in[c * N_LEN + n] * stem_w[c * 64 + d];
    xT[(n + 1) * 64 + d] = (_Float16)s;
  }
  if (tid < 64) { xT[tid] = (_Float16)0.f; xT[513 * 64 + tid] = (_Float16)0.f; }
  __syncthreads();

  // ---- 5 residual conv blocks ----
  for (int blk = 0; blk < 5; ++blk) {
    const float* cw = args.cp[blk].w;                 // [o][i][t] (OIH)
    for (int idx = tid; idx < 64 * 192; idx += 256) { // repack -> wA[d][t*64+c]
      int d = idx / 192, k = idx % 192;
      int t = k >> 6, c = k & 63;
      wA[idx] = (_Float16)cw[d * 192 + c * 3 + t];
    }
    if (tid < 64) { gG[tid] = args.cp[blk].g[tid]; gB[tid] = args.cp[blk].b[tid]; }
    __syncthreads();

    // WMMA GEMM: 4 M-tiles x 32 N-tiles, 6 K-steps of 32. 16 C-tiles per wave.
    for (int tt = 0; tt < 16; ++tt) {
      int tile = w * 16 + tt;
      int mt = tile & 3, nt = tile >> 2;
      v8f acc = {};
#pragma unroll
      for (int kb = 0; kb < 6; ++kb) {
        union { v16h v; _Float16 h[16]; float4 f4[2]; } af, bf;
        // A (16x32 f16): lane half l>=16 takes K+8; halfs e<8 -> K=k0+e,
        // e>=8 -> K=k0+16+(e-8)  => two 16B chunks, 16 halfs apart.
        int mrow = mt * 16 + (lane & 15);
        int k0   = kb * 32 + ((lane >> 4) << 3);
        const _Float16* pa = &wA[mrow * 192 + k0];
        af.f4[0] = *(const float4*)(pa);
        af.f4[1] = *(const float4*)(pa + 16);
        // B (32x16 f16): lane n = nt*16+(lane&15); K-local = 8*(lane>=16)+e (e<8), 16+8*(lane>=16)+(e-8) (e>=8).
        // k = kb*32 + klocal -> tap t = kb>>1, channel c0 = (kb&1)*32.
        int n  = nt * 16 + (lane & 15);
        int t  = kb >> 1;
        int c0 = ((kb & 1) << 5) + ((lane >> 4) << 3);
        const _Float16* pb = &xT[(n + t) * 64 + c0];  // halo: row n+t == x[n+t-1]
        bf.f4[0] = *(const float4*)(pb);
        bf.f4[1] = *(const float4*)(pb + 16);
        acc = __builtin_amdgcn_wmma_f32_16x16x32_f16(
            false, af.v, false, bf.v, (short)0, acc, false, false);
      }
      // C/D layout: vgpr i -> M = i + 8*(lane>=16), N = lane&15
      int nn    = nt * 16 + (lane & 15);
      int mbase = mt * 16 + ((lane >> 4) << 3);
#pragma unroll
      for (int i = 0; i < 8; ++i) yT[nn * 64 + mbase + i] = (_Float16)acc[i];
    }
    __syncthreads();

    // GroupNorm stats: 8 groups of 8 channels x 512; wave w owns group w.
    float s = 0.f, ss = 0.f;
    for (int n = lane; n < N_LEN; n += 32) {
#pragma unroll
      for (int c = 0; c < 8; ++c) {
        float v = (float)yT[n * 64 + w * 8 + c];
        s += v; ss += v * v;
      }
    }
    s = wred_sum(s); ss = wred_sum(ss);
    if (lane == 0) {
      float mu  = s * (1.f / 4096.f);
      float var = ss * (1.f / 4096.f) - mu * mu;
      gstat[w * 2] = mu; gstat[w * 2 + 1] = rsqrtf(var + 1e-5f);
    }
    __syncthreads();

    // x += gelu(groupnorm(y))
    for (int idx = tid; idx < N_LEN * 64; idx += 256) {
      int d = idx & 63, g = d >> 3;
      float v  = (float)yT[idx];
      float xn = (v - gstat[g * 2]) * gstat[g * 2 + 1] * gG[d] + gB[d];
      float r  = (float)xT[idx + 64];
      xT[idx + 64] = (_Float16)(r + gelu_exact(xn));
    }
    __syncthreads();
  }

  // ---- Average pool over N ----
  {
    int d = tid & 63, q = tid >> 6;
    float s = 0.f;
    for (int n = q; n < N_LEN; n += 4) s += (float)xT[(n + 1) * 64 + d];
    pool[q * 64 + d] = s;
    __syncthreads();
    if (tid < 64) {
      float t = pool[tid] + pool[64 + tid] + pool[128 + tid] + pool[192 + tid];
      conv_emb[(size_t)m * 64 + tid] = t * (1.f / 512.f);
    }
  }
}

// ---------------------------------------------------------------------------
// K2: edge merge  e = gelu(LN(concat(conv_emb, type_emb) @ W + b))
// ---------------------------------------------------------------------------
__global__ __launch_bounds__(64)
void k2_edge_merge(const float* __restrict__ conv_emb,
                   const int* __restrict__ edge_types,
                   const float* __restrict__ type_emb,
                   const float* __restrict__ Wm, const float* __restrict__ bm,
                   const float* __restrict__ lng, const float* __restrict__ lnb,
                   float* __restrict__ e_out)
{
  __shared__ float row[64];
  int mrow = blockIdx.x;
  int d = threadIdx.x;
  int ty = edge_types[mrow];
  const float* ce = conv_emb + (size_t)mrow * 64;
  const float* te = type_emb + (size_t)ty * 64;
  float a = bm[d];
  for (int c = 0; c < 64; ++c) a += ce[c] * Wm[c * 64 + d];
  for (int c = 0; c < 64; ++c) a += te[c] * Wm[(64 + c) * 64 + d];
  row[d] = a;
  __syncthreads();
  float mu = 0.f;
  for (int c = 0; c < 64; ++c) mu += row[c];
  mu *= (1.f / 64.f);
  float var = 0.f;
  for (int c = 0; c < 64; ++c) { float q = row[c] - mu; var += q * q; }
  var *= (1.f / 64.f);
  float xn = (a - mu) * rsqrtf(var + 1e-5f) * lng[d] + lnb[d];
  e_out[(size_t)mrow * 64 + d] = gelu_exact(xn);
}

// ---------------------------------------------------------------------------
// K3: one transformer layer per call; one workgroup per batch element b.
// ---------------------------------------------------------------------------
struct AttnP {
  const float *Wq, *bq, *Wk, *bk, *Wv, *bv, *Wo, *bo;
  const float *sbias, *ln1g, *ln1b;
  const float *W1, *b1, *W2, *b2, *ln2g, *ln2b;
};

__global__ __launch_bounds__(256)
void k3_attn(float* __restrict__ e_buf, const int* __restrict__ struct_rel, AttnP p)
{
  __shared__ float    E0[E_EDGES * 64];
  __shared__ _Float16 Qh[E_EDGES * 64], Kh[E_EDGES * 64], Vh[E_EDGES * 64];
  __shared__ float    AO[E_EDGES * 64];
  __shared__ float    swrow[8 * E_EDGES];
  __shared__ float    ffh[8 * 256];

  int b = blockIdx.x, tid = threadIdx.x, lane = tid & 31, w = tid >> 5;
  float* eb = e_buf + (size_t)b * E_EDGES * 64;
  const int* rel = struct_rel + (size_t)b * E_EDGES * E_EDGES;

  for (int idx = tid; idx < E_EDGES * 64; idx += 256) E0[idx] = eb[idx];
  __syncthreads();

  // QKV projections
  for (int idx = tid; idx < E_EDGES * 64; idx += 256) {
    int i = idx >> 6, d = idx & 63;
    float q = p.bq[d], k = p.bk[d], v = p.bv[d];
    const float* x = &E0[i * 64];
    for (int c = 0; c < 64; ++c) {
      float xv = x[c];
      q += xv * p.Wq[c * 64 + d];
      k += xv * p.Wk[c * 64 + d];
      v += xv * p.Wv[c * 64 + d];
    }
    Qh[idx] = (_Float16)q; Kh[idx] = (_Float16)k; Vh[idx] = (_Float16)v;
  }
  __syncthreads();

  // Attention: one (head, query-row) task per wave iteration.
  for (int task = w; task < H_HEADS * E_EDGES; task += 8) {
    int h = task & 3, qi = task >> 2;
    float sc[5]; int cnt = 0; float mx = -3.0e38f;
    for (int kj = lane; kj < E_EDGES; kj += 32) {
      float s = 0.f;
#pragma unroll
      for (int dd = 0; dd < 16; ++dd)
        s += (float)Qh[qi * 64 + h * 16 + dd] * (float)Kh[kj * 64 + h * 16 + dd];
      s = s * 0.25f + p.sbias[rel[qi * E_EDGES + kj] * H_HEADS + h];
      sc[cnt++] = s; mx = fmaxf(mx, s);
    }
    mx = wred_max(mx);
    float sum = 0.f; cnt = 0;
    for (int kj = lane; kj < E_EDGES; kj += 32) {
      float e2 = __expf(sc[cnt] - mx); sc[cnt] = e2; sum += e2; ++cnt;
    }
    sum = wred_sum(sum);
    float inv = 1.f / sum; cnt = 0;
    for (int kj = lane; kj < E_EDGES; kj += 32) swrow[w * E_EDGES + kj] = sc[cnt++] * inv;
    // out = w @ V  (lanes split keys in halves, dd = lane&15)
    int dd = lane & 15, half = lane >> 4;
    float o = 0.f;
    for (int kj = half; kj < E_EDGES; kj += 2)
      o += swrow[w * E_EDGES + kj] * (float)Vh[kj * 64 + h * 16 + dd];
    o += __shfl_xor(o, 16, 32);
    if (lane < 16) AO[qi * 64 + h * 16 + dd] = o;
  }
  __syncthreads();

  // out-proj + residual + LN1 (wave per row; each lane owns 2 dims)
  for (int r = w; r < E_EDGES; r += 8) {
    int d0 = lane, d1 = lane + 32;
    float a0 = p.bo[d0], a1 = p.bo[d1];
    for (int c = 0; c < 64; ++c) {
      float x = AO[r * 64 + c];
      a0 += x * p.Wo[c * 64 + d0];
      a1 += x * p.Wo[c * 64 + d1];
    }
    a0 += E0[r * 64 + d0]; a1 += E0[r * 64 + d1];
    float mu = wred_sum(a0 + a1) * (1.f / 64.f);
    float q0 = a0 - mu, q1 = a1 - mu;
    float rs = rsqrtf(wred_sum(q0 * q0 + q1 * q1) * (1.f / 64.f) + 1e-5f);
    E0[r * 64 + d0] = q0 * rs * p.ln1g[d0] + p.ln1b[d0];
    E0[r * 64 + d1] = q1 * rs * p.ln1g[d1] + p.ln1b[d1];
  }
  __syncthreads();

  // FFN 64->256->64 + residual + LN2
  for (int r = w; r < E_EDGES; r += 8) {
    for (int hh = lane; hh < 256; hh += 32) {
      float a = p.b1[hh];
      for (int c = 0; c < 64; ++c) a += E0[r * 64 + c] * p.W1[c * 256 + hh];
      ffh[w * 256 + hh] = gelu_exact(a);
    }
    int d0 = lane, d1 = lane + 32;
    float a0 = p.b2[d0], a1 = p.b2[d1];
    for (int c = 0; c < 256; ++c) {
      float x = ffh[w * 256 + c];
      a0 += x * p.W2[c * 64 + d0];
      a1 += x * p.W2[c * 64 + d1];
    }
    a0 += E0[r * 64 + d0]; a1 += E0[r * 64 + d1];
    float mu = wred_sum(a0 + a1) * (1.f / 64.f);
    float q0 = a0 - mu, q1 = a1 - mu;
    float rs = rsqrtf(wred_sum(q0 * q0 + q1 * q1) * (1.f / 64.f) + 1e-5f);
    E0[r * 64 + d0] = q0 * rs * p.ln2g[d0] + p.ln2b[d0];
    E0[r * 64 + d1] = q1 * rs * p.ln2g[d1] + p.ln2b[d1];
  }
  __syncthreads();
  for (int idx = tid; idx < E_EDGES * 64; idx += 256) eb[idx] = E0[idx];
}

// ---------------------------------------------------------------------------
// K4: edge head + node gather/merge/head.  One block per batch element.
// IDX_VX[v]=11v, IDX_VY[v]=11v+1, IDX_XV[v]=v-1, IDX_YV[v]=10+v  (v=j+2)
// ---------------------------------------------------------------------------
__global__ __launch_bounds__(256)
void k4_heads(const float* __restrict__ e_buf,
              const float* __restrict__ Whe, const float* __restrict__ bhe,
              const float* __restrict__ Wnm, const float* __restrict__ bnm,
              const float* __restrict__ lng, const float* __restrict__ lnb,
              const float* __restrict__ Wnh, const float* __restrict__ bnh,
              float* __restrict__ out)
{
  __shared__ float nh[8 * 64];
  int b = blockIdx.x, tid = threadIdx.x, lane = tid & 31, w = tid >> 5;
  const float* eb = e_buf + (size_t)b * E_EDGES * 64;

  // edge logits [132,2]
  for (int t = tid; t < E_EDGES * 2; t += 256) {
    int i = t >> 1, o = t & 1;
    float a = bhe[o];
    for (int c = 0; c < 64; ++c) a += eb[i * 64 + c] * Whe[c * 2 + o];
    out[(size_t)b * E_EDGES * 2 + t] = a;
  }

  // node logits [10,8]
  for (int nj = w; nj < 10; nj += 8) {
    int v = nj + 2;
    const float* seg[4] = { eb + (11 * v) * 64, eb + (11 * v + 1) * 64,
                            eb + (v - 1) * 64,  eb + (10 + v) * 64 };
    int hd0 = lane, hd1 = lane + 32;
    float h0 = bnm[hd0], h1 = bnm[hd1];
#pragma unroll
    for (int sgi = 0; sgi < 4; ++sgi) {
      const float* sp = seg[sgi];
      for (int c = 0; c < 64; ++c) {
        float x = sp[c];
        h0 += x * Wnm[(sgi * 64 + c) * 64 + hd0];
        h1 += x * Wnm[(sgi * 64 + c) * 64 + hd1];
      }
    }
    float mu = wred_sum(h0 + h1) * (1.f / 64.f);
    float q0 = h0 - mu, q1 = h1 - mu;
    float rs = rsqrtf(wred_sum(q0 * q0 + q1 * q1) * (1.f / 64.f) + 1e-5f);
    nh[w * 64 + hd0] = gelu_exact(q0 * rs * lng[hd0] + lnb[hd0]);
    nh[w * 64 + hd1] = gelu_exact(q1 * rs * lng[hd1] + lnb[hd1]);
    if (lane < 8) {
      float a = bnh[lane];
      for (int c = 0; c < 64; ++c) a += nh[w * 64 + c] * Wnh[c * 8 + lane];
      out[(size_t)(E_EDGES * 2 * B_SZ) + (size_t)b * 80 + nj * 8 + lane] = a;
    }
  }
}

// ---------------------------------------------------------------------------
extern "C" void kernel_launch(void* const* d_in, const int* in_sizes, int n_in,
                              void* d_out, int out_size, void* d_ws, size_t ws_size,
                              hipStream_t stream)
{
  (void)in_sizes; (void)n_in; (void)out_size; (void)ws_size;
  // setup_inputs insertion order, params flattened in insertion order:
  const float* edge_data  = (const float*)d_in[0];
  const int*   edge_types = (const int*)d_in[1];
  /* d_in[2] = edge_mask (all true) -- unused */
  const int*   struct_rel = (const int*)d_in[3];
  const float* stem_w = (const float*)d_in[4];
  const float* stem_b = (const float*)d_in[5];
  K1Args k1a;
  for (int i = 0; i < 5; ++i) {
    k1a.cp[i].w = (const float*)d_in[6 + 3 * i];
    k1a.cp[i].g = (const float*)d_in[7 + 3 * i];
    k1a.cp[i].b = (const float*)d_in[8 + 3 * i];
  }
  const float* type_emb = (const float*)d_in[21];
  const float* em_w  = (const float*)d_in[22];
  const float* em_b  = (const float*)d_in[23];
  const float* em_g  = (const float*)d_in[24];
  const float* em_bb = (const float*)d_in[25];
  AttnP ap[2];
  for (int l = 0; l < 2; ++l) {
    int o = 26 + l * 17;
    ap[l].Wq = (const float*)d_in[o + 0];  ap[l].bq = (const float*)d_in[o + 1];
    ap[l].Wk = (const float*)d_in[o + 2];  ap[l].bk = (const float*)d_in[o + 3];
    ap[l].Wv = (const float*)d_in[o + 4];  ap[l].bv = (const float*)d_in[o + 5];
    ap[l].Wo = (const float*)d_in[o + 6];  ap[l].bo = (const float*)d_in[o + 7];
    ap[l].sbias = (const float*)d_in[o + 8];
    ap[l].ln1g = (const float*)d_in[o + 9];  ap[l].ln1b = (const float*)d_in[o + 10];
    ap[l].W1 = (const float*)d_in[o + 11];   ap[l].b1 = (const float*)d_in[o + 12];
    ap[l].W2 = (const float*)d_in[o + 13];   ap[l].b2 = (const float*)d_in[o + 14];
    ap[l].ln2g = (const float*)d_in[o + 15]; ap[l].ln2b = (const float*)d_in[o + 16];
  }
  const float* eh_w  = (const float*)d_in[60];
  const float* eh_b  = (const float*)d_in[61];
  const float* nm_w  = (const float*)d_in[62];
  const float* nm_b  = (const float*)d_in[63];
  const float* nm_g  = (const float*)d_in[64];
  const float* nm_bb = (const float*)d_in[65];
  const float* nh_w  = (const float*)d_in[66];
  const float* nh_b  = (const float*)d_in[67];

  float* conv_emb = (float*)d_ws;                              // 2112*64 f32
  float* e_buf    = (float*)((char*)d_ws + 544768);            // 16*132*64 f32

  k1_stem_conv_pool<<<M_TOT, 256, 0, stream>>>(edge_data, stem_w, stem_b, k1a, conv_emb);
  k2_edge_merge<<<M_TOT, 64, 0, stream>>>(conv_emb, edge_types, type_emb,
                                          em_w, em_b, em_g, em_bb, e_buf);
  k3_attn<<<B_SZ, 256, 0, stream>>>(e_buf, struct_rel, ap[0]);
  k3_attn<<<B_SZ, 256, 0, stream>>>(e_buf, struct_rel, ap[1]);
  k4_heads<<<B_SZ, 256, 0, stream>>>(e_buf, eh_w, eh_b, nm_w, nm_b, nm_g, nm_bb,
                                     nh_w, nh_b, (float*)d_out);
}
